// GE2ELoss_87540023427916
// MI455X (gfx1250) — compile-verified, matrix operand order checked by reading, not run on hardware
//
#include <hip/hip_runtime.h>
#include <hip/hip_bf16.h>
#include <math.h>
#include <stdint.h>

typedef __attribute__((ext_vector_type(16))) _Float16 v16h;
typedef __attribute__((ext_vector_type(8)))  float    v8f;
typedef __attribute__((ext_vector_type(4)))  _Float16 v4h;

// GCC-style vector to exactly match the builtin's parameter pointee type:
//   __attribute__((__vector_size__(4 * sizeof(int)))) int
typedef int i32x4_vs __attribute__((vector_size(16)));

#define N_SPK 1024
#define M_UTT 16
#define D_DIM 1024
#define NM    (N_SPK * M_UTT)
#define EPS   1e-8f

// ---------------------------------------------------------------------------
// gfx1250 async global->LDS copy (ASYNCcnt-tracked DMA path) + wait.
// Probe-confirmed: __builtin_amdgcn_global_load_async_to_lds_b128 exists and
// takes (AS1 int4* src, AS3 int4* lds_dst, imm offset, imm cpol).
// LDS operand built from the low 32 bits of a generic shared-memory pointer
// (flat-aperture truncation rule); AS pointers built via int-to-pointer casts.
// ---------------------------------------------------------------------------
#if __has_builtin(__builtin_amdgcn_global_load_async_to_lds_b128)
#define ASYNC_COPY_B128(lptr, gptr)                                            \
  __builtin_amdgcn_global_load_async_to_lds_b128(                              \
      (__attribute__((address_space(1))) i32x4_vs*)(uintptr_t)(gptr),          \
      (__attribute__((address_space(3))) i32x4_vs*)(unsigned)(uintptr_t)(lptr),\
      0, 0)
#else
#define ASYNC_COPY_B128(lptr, gptr)                                            \
  asm volatile("global_load_async_to_lds_b128 %0, %1, off"                     \
               :: "v"((unsigned)(uintptr_t)(lptr)),                            \
                  "v"((unsigned long long)(uintptr_t)(gptr))                   \
               : "memory")
#endif

#if __has_builtin(__builtin_amdgcn_s_wait_asynccnt)
#define WAIT_ASYNC(n) __builtin_amdgcn_s_wait_asynccnt(n)
#else
#define WAIT_ASYNC(n) asm volatile("s_wait_asynccnt %0" :: "i"(n) : "memory")
#endif

// ---------------------------------------------------------------------------
// Block-wide sum over 256 threads (8 wave32s), broadcast result to all threads
// ---------------------------------------------------------------------------
__device__ __forceinline__ float blockReduce256(float v, float* red) {
  const int lane = threadIdx.x & 31;
  const int wid  = threadIdx.x >> 5;
#pragma unroll
  for (int off = 16; off > 0; off >>= 1) v += __shfl_down(v, off, 32);
  __syncthreads();               // protect red[] from previous round's readers
  if (lane == 0) red[wid] = v;
  __syncthreads();
  float r = red[0];
#pragma unroll
  for (int i = 1; i < 8; ++i) r += red[i];
  return r;
}

// ---------------------------------------------------------------------------
// Kernel 1: per-speaker preprocessing.
//   Ehat[j*M+m][d]  = f16 normalized embedding
//   Chat[j][d]      = f16 normalized full centroid
//   diag[j*M+m]     = cos(e_hat[j,m], excl_hat[j,m])   (f32)
// ---------------------------------------------------------------------------
__global__ void ge2e_preprocess(const float* __restrict__ E,
                                _Float16* __restrict__ Ehat,
                                _Float16* __restrict__ Chat,
                                float* __restrict__ diag) {
  __shared__ float tile[M_UTT * D_DIM];   // 64 KB: this speaker's raw f32 tile
  __shared__ float red[8];
  const int j  = blockIdx.x;
  const int t  = threadIdx.x;             // 0..255
  const float4* Ej = (const float4*)(E + (size_t)j * M_UTT * D_DIM);

  float4 sumv = make_float4(0.f, 0.f, 0.f, 0.f);
  float rnorm[M_UTT];

#pragma unroll 1
  for (int m = 0; m < M_UTT; ++m) {
    float4 v = Ej[m * (D_DIM / 4) + t];
    ((float4*)(tile + m * D_DIM))[t] = v;
    sumv.x += v.x; sumv.y += v.y; sumv.z += v.z; sumv.w += v.w;
    float sq = v.x * v.x + v.y * v.y + v.z * v.z + v.w * v.w;
    float ns = blockReduce256(sq, red);
    rnorm[m] = 1.0f / fmaxf(sqrtf(ns), EPS);
  }

  float4 c;
  c.x = sumv.x * (1.0f / M_UTT); c.y = sumv.y * (1.0f / M_UTT);
  c.z = sumv.z * (1.0f / M_UTT); c.w = sumv.w * (1.0f / M_UTT);
  float cns = blockReduce256(c.x * c.x + c.y * c.y + c.z * c.z + c.w * c.w, red);
  float rc  = 1.0f / fmaxf(sqrtf(cns), EPS);
  {
    v4h ch;
    ch[0] = (_Float16)(c.x * rc); ch[1] = (_Float16)(c.y * rc);
    ch[2] = (_Float16)(c.z * rc); ch[3] = (_Float16)(c.w * rc);
    ((v4h*)(Chat + (size_t)j * D_DIM))[t] = ch;
  }

#pragma unroll 1
  for (int m = 0; m < M_UTT; ++m) {
    float4 v = ((float4*)(tile + m * D_DIM))[t];
    const float rn = rnorm[m];
    v4h eh;
    eh[0] = (_Float16)(v.x * rn); eh[1] = (_Float16)(v.y * rn);
    eh[2] = (_Float16)(v.z * rn); eh[3] = (_Float16)(v.w * rn);
    ((v4h*)(Ehat + ((size_t)j * M_UTT + m) * D_DIM))[t] = eh;

    float4 ex;
    ex.x = (sumv.x - v.x) * (1.0f / (M_UTT - 1));
    ex.y = (sumv.y - v.y) * (1.0f / (M_UTT - 1));
    ex.z = (sumv.z - v.z) * (1.0f / (M_UTT - 1));
    ex.w = (sumv.w - v.w) * (1.0f / (M_UTT - 1));
    float exsq = ex.x * ex.x + ex.y * ex.y + ex.z * ex.z + ex.w * ex.w;
    float dve  = v.x * ex.x + v.y * ex.y + v.z * ex.z + v.w * ex.w;
    float EX = blockReduce256(exsq, red);
    float DV = blockReduce256(dve,  red);
    if (t == 0) {
      float rex = 1.0f / fmaxf(sqrtf(EX), EPS);
      diag[j * M_UTT + m] = DV * rn * rex;
    }
  }
}

// ---------------------------------------------------------------------------
// Kernel 2: zero rowsum accumulators
// ---------------------------------------------------------------------------
__global__ void ge2e_zero(float* __restrict__ p) {
  p[blockIdx.x * 256 + threadIdx.x] = 0.0f;
}

// ---------------------------------------------------------------------------
// Kernel 3: WMMA GEMM (Ehat x Chat^T) fused with exp-sum epilogue.
// Block = 128 threads (4 waves): 64 rows x 64 cols. K loop in steps of 64.
// B fragments are DMA'd global->LDS with the gfx1250 async path, double
// buffered so iteration i+1's transfers overlap iteration i's 8 WMMAs.
// Diagonal column (k == speaker) contributes 0 here; added in finalize.
// ---------------------------------------------------------------------------
__global__ void __launch_bounds__(128)
ge2e_gemm(const _Float16* __restrict__ Ehat,
          const _Float16* __restrict__ Chat,
          float* __restrict__ rowsum,
          const float* __restrict__ wp,
          const float* __restrict__ bp) {
  // [buf][ (tile c, subchunk q) ][ lane ][ 16 halves ] : 2 x 8 KB
  __shared__ _Float16 ldsB[2][4096];

  const int tid = threadIdx.x;
  const int wv  = tid >> 5;               // wave 0..3 -> 16-row strip
  const int L   = tid & 31;               // lane
  const int n   = L & 15;                 // column-within-tile / row-within-A
  const int hi  = L >> 4;                 // K-half selector per ISA A/B layout

  const int colBase = blockIdx.x * 64;
  const int rowBase = blockIdx.y * 64;
  const int rBase   = rowBase + wv * 16;  // 16 rows of one speaker
  const int arow    = rBase + n;

  v8f acc[4];
#pragma unroll
  for (int c = 0; c < 4; ++c) acc[c] = v8f{};

  // cooperative-staging role: this thread stages col-tile `wv`, lane slot L
  const int ccol = colBase + wv * 16 + n;
  const _Float16* gbBase = Chat + (size_t)ccol * D_DIM + 8 * hi;
  const _Float16* gaBase = Ehat + (size_t)arow * D_DIM + 8 * hi;

  union Frag { v16h h; uint4 u[2]; };

  // Issue 4 async b128 DMAs staging K-chunks [k0,k0+32) and [k0+32,k0+64)
  auto stage = [&](int k0, int buf) {
#pragma unroll
    for (int q = 0; q < 2; ++q) {
      const _Float16* g = gbBase + k0 + q * 32;
      _Float16* l = &ldsB[buf][((wv * 2 + q) * 32 + L) * 16];
      ASYNC_COPY_B128(l,     g);
      ASYNC_COPY_B128(l + 8, g + 16);
    }
  };

  stage(0, 0);
  int cur = 0;

#pragma unroll 1
  for (int k0 = 0; k0 < D_DIM; k0 += 64) {
    // kick off next buffer's DMA, then wait for current buffer's 4 loads
    // (async loads complete in order per wave)
    if (k0 + 64 < D_DIM) {
      stage(k0 + 64, cur ^ 1);
      WAIT_ASYNC(4);
    } else {
      WAIT_ASYNC(0);
    }
    __syncthreads();              // current buffer visible to all waves

#pragma unroll
    for (int q = 0; q < 2; ++q) {
      Frag A;                     // A fragment straight from global (L2-hot)
      const _Float16* ga = gaBase + k0 + q * 32;
      A.u[0] = *(const uint4*)ga;
      A.u[1] = *(const uint4*)(ga + 16);
#pragma unroll
      for (int c = 0; c < 4; ++c) {
        Frag B;
        const uint4* lr = (const uint4*)&ldsB[cur][((c * 2 + q) * 32 + L) * 16];
        B.u[0] = lr[0];
        B.u[1] = lr[1];
        acc[c] = __builtin_amdgcn_wmma_f32_16x16x32_f16(
            false, A.h, false, B.h, (short)0, acc[c], false, false);
      }
    }
    __syncthreads();              // all waves done reading before re-stage
    cur ^= 1;
  }

  // Epilogue: logits = w*sim + b; partial sum of exp over this col range,
  // skipping the diagonal column (handled exactly in finalize).
  const float w = wp[0], b = bp[0];
  const int jr = rBase >> 4;              // speaker index of this 16-row strip

  float s[8];
#pragma unroll
  for (int i = 0; i < 8; ++i) s[i] = 0.0f;
#pragma unroll
  for (int c = 0; c < 4; ++c) {
    const int k = colBase + c * 16 + n;
    const float keep = (k == jr) ? 0.0f : 1.0f;
#pragma unroll
    for (int i = 0; i < 8; ++i)
      s[i] += keep * __expf(w * acc[c][i] + b);
  }
  // reduce across each 16-lane half (rows i and i+8 live in separate halves)
#pragma unroll
  for (int i = 0; i < 8; ++i) {
#pragma unroll
    for (int off = 1; off < 16; off <<= 1)
      s[i] += __shfl_xor(s[i], off, 32);
    if (n == 0)
      atomicAdd(&rowsum[rBase + i + 8 * hi], s[i]);
  }
}

// ---------------------------------------------------------------------------
// Kernel 4: loss = mean over rows of log(rowsum + exp(dl)) - dl,
//           dl = w*diag + b  (true logit, also the diagonal's exp term)
// ---------------------------------------------------------------------------
__global__ void ge2e_finalize(const float* __restrict__ rowsum,
                              const float* __restrict__ diag,
                              const float* __restrict__ wp,
                              const float* __restrict__ bp,
                              float* __restrict__ out) {
  __shared__ float red[8];
  const float w = wp[0], b = bp[0];
  float local = 0.0f;
  for (int r = threadIdx.x; r < NM; r += 256) {
    float dl = w * diag[r] + b;
    local += logf(rowsum[r] + __expf(dl)) - dl;
  }
  float tot = blockReduce256(local, red);
  if (threadIdx.x == 0) out[0] = tot * (1.0f / (float)NM);
}

// ---------------------------------------------------------------------------
extern "C" void kernel_launch(void* const* d_in, const int* in_sizes, int n_in,
                              void* d_out, int out_size, void* d_ws, size_t ws_size,
                              hipStream_t stream) {
  const float* E  = (const float*)d_in[0];
  const float* wp = (const float*)d_in[1];
  const float* bp = (const float*)d_in[2];
  float* out = (float*)d_out;

  char* ws = (char*)d_ws;
  _Float16* Ehat = (_Float16*)ws;                                   // 32 MB
  _Float16* Chat = (_Float16*)(ws + (size_t)NM * D_DIM * 2);        //  2 MB
  float*    diag = (float*)(ws + (size_t)NM * D_DIM * 2
                               + (size_t)N_SPK * D_DIM * 2);        // 64 KB
  float* rowsum  = diag + NM;                                       // 64 KB

  ge2e_preprocess<<<N_SPK, 256, 0, stream>>>(E, Ehat, Chat, diag);
  ge2e_zero<<<NM / 256, 256, 0, stream>>>(rowsum);
  ge2e_gemm<<<dim3(N_SPK / 64, NM / 64), 128, 0, stream>>>(Ehat, Chat, rowsum, wp, bp);
  ge2e_finalize<<<1, 256, 0, stream>>>(rowsum, diag, wp, bp, out);
}
